// MultiHeadAttention_88295937671479
// MI455X (gfx1250) — compile-verified, matrix operand order checked by reading, not run on hardware
//
#include <hip/hip_runtime.h>
#include <hip/hip_bf16.h>
#include <cstdint>

// ---------------------------------------------------------------------------
// Multi-head attention for MI455X (gfx1250, wave32, WMMA + TDM).
//   1) convert x, Wq, Wk, Wv, Wo  f32 -> bf16   (workspace)
//   2) Q/K/V = x W^T + b  (bf16 WMMA GEMM, TDM-staged, double-buffered LDS)
//   3) flash attention per (b, h, 64 q rows): logits/64, online softmax
//   4) out = attn Wo^T + bo  (fp32 output)
// ---------------------------------------------------------------------------

typedef __attribute__((ext_vector_type(16))) __bf16       v16bf;
typedef __attribute__((ext_vector_type(8)))  float        v8f;
typedef __attribute__((ext_vector_type(4)))  unsigned int u32x4;
typedef __attribute__((ext_vector_type(8)))  int          i32x8;
typedef __attribute__((ext_vector_type(4)))  int          i32x4;

union FragU {
    uint4  q[2];   // two 16-byte chunks
    v16bf  v;      // one WMMA 16-bit operand fragment (8 VGPRs)
};

__device__ __forceinline__ unsigned short bfbits(float f) {
    unsigned u = __builtin_bit_cast(unsigned, f);
    return (unsigned short)((u + 0x7FFFu + ((u >> 16) & 1u)) >> 16);  // RNE
}
__device__ __forceinline__ __bf16 f2bf(float f) {
    unsigned short s = bfbits(f);
    return __builtin_bit_cast(__bf16, s);
}

// --------------------------- TDM availability ------------------------------
#if defined(__has_builtin)
#if __has_builtin(__builtin_amdgcn_tensor_load_to_lds) && \
    __has_builtin(__builtin_amdgcn_s_wait_tensorcnt)
#define USE_TDM 1
#endif
#endif
#ifndef USE_TDM
#define USE_TDM 0
#endif

#if USE_TDM
// Build a D# for a dense 2D bf16 tile (rows x rowlen elems, row stride
// `stride_elems`) and issue TENSOR_LOAD_TO_LDS.  Per CDNA5 ISA:
//  group0: [0]=count=1, [1]=lds_addr, [2]=global_addr[31:0],
//          [3]=global_addr[56:32] | type=2<<30
//  group1: [0]=data_size(2B)<<16, tensor_dim0/1, tile_dim0/1, dim0 stride
__device__ __forceinline__ void tdm_load_2d_bf16(
    const void* gaddr, uint32_t lds_byte_off,
    uint32_t rows, uint32_t rowlen, uint32_t stride_elems)
{
    uint64_t ga = (uint64_t)(uintptr_t)gaddr;
    u32x4 g0;
    g0[0] = 1u;                                   // count=1, user descriptor
    g0[1] = lds_byte_off;                         // LDS destination (bytes)
    g0[2] = (uint32_t)(ga & 0xFFFFFFFFu);
    g0[3] = (uint32_t)((ga >> 32) & 0x01FFFFFFu) | (2u << 30);  // type=2
    i32x8 g1;
    g1[0] = (int)(1u << 16);                      // data_size = 2 bytes
    g1[1] = (int)((rowlen & 0xFFFFu) << 16);      // tensor_dim0[15:0]
    g1[2] = (int)(((rowlen >> 16) & 0xFFFFu) |    // tensor_dim0[31:16]
                  ((rows & 0xFFFFu) << 16));      // tensor_dim1[15:0]
    g1[3] = (int)(((rows >> 16) & 0xFFFFu) |      // tensor_dim1[31:16]
                  ((rowlen & 0xFFFFu) << 16));    // tile_dim0
    g1[4] = (int)(rows & 0xFFFFu);                // tile_dim1 (tile_dim2 = 0)
    g1[5] = (int)stride_elems;                    // tensor_dim0_stride[31:0]
    g1[6] = 0;                                    // stride hi / dim1 stride lo
    g1[7] = 0;
    i32x4 z4 = {0, 0, 0, 0};
#if defined(__clang_major__) && (__clang_major__ >= 23)
    i32x8 z8 = {0, 0, 0, 0, 0, 0, 0, 0};
    __builtin_amdgcn_tensor_load_to_lds(g0, g1, z4, z4, z8, 0);
#else
    __builtin_amdgcn_tensor_load_to_lds(g0, g1, z4, z4, 0);
#endif
}
__device__ __forceinline__ uint32_t lds_off(const void* p) {
    // LDS aperture: generic addr low 32 bits are the LDS byte offset.
    return (uint32_t)(uintptr_t)p;
}
#endif  // USE_TDM

// ---------------------------------------------------------------------------
// f32 -> bf16 conversion, 4 elements / thread
// ---------------------------------------------------------------------------
__global__ __launch_bounds__(256) void cvt_f32_bf16(
    const float* __restrict__ in, __bf16* __restrict__ out, int n)
{
    int i = (blockIdx.x * blockDim.x + threadIdx.x) * 4;
    if (i + 3 < n) {
        float4 f = *(const float4*)(in + i);
        ushort4 o;
        o.x = bfbits(f.x); o.y = bfbits(f.y);
        o.z = bfbits(f.z); o.w = bfbits(f.w);
        *(ushort4*)(out + i) = o;
    } else {
        for (int j = i; j < n; ++j) out[j] = f2bf(in[j]);
    }
}

// ---------------------------------------------------------------------------
// Tiled bf16 WMMA GEMM:  C[M,N] = A[M,K] * W[N,K]^T + bias[N]
// Workgroup tile 128x128, 8 waves; wave = 32x64 patch (2x4 WMMA accums).
// K staged via LDS (TDM double-buffered when available), 32/iter.
// ---------------------------------------------------------------------------
#define GTM 128
#define GTN 128
#define GTK 32

__global__ __launch_bounds__(256) void wmma_gemm_bf16(
    const __bf16* __restrict__ A, const __bf16* __restrict__ W,
    const float* __restrict__ bias, void* __restrict__ Cout,
    int M, int N, int K, int out_is_bf16)
{
#if USE_TDM
    __shared__ __bf16 As[2][GTM][GTK];   // 16 KB, double buffered
    __shared__ __bf16 Bs[2][GTN][GTK];   // 16 KB
#else
    __shared__ __bf16 As[1][GTM][GTK];   // 8 KB
    __shared__ __bf16 Bs[1][GTN][GTK];   // 8 KB
#endif

    const int tid  = threadIdx.x;
    const int lane = tid & 31;
    const int wave = tid >> 5;        // 0..7
    const int wm   = wave >> 1;       // 0..3 (M dir, 32 rows each)
    const int wn   = wave & 1;        // 0..1 (N dir, 64 cols each)
    const int half = lane >> 4;       // 0/1
    const int l16  = lane & 15;

    const int tileM = blockIdx.x * GTM;
    const int tileN = blockIdx.y * GTN;
    const int nIter = K / GTK;

    v8f acc[2][4] = {};

#if USE_TDM
    if (wave == 0) {   // uniform branch: wave 0 drives the TDM
        tdm_load_2d_bf16(A + (size_t)tileM * K, lds_off(&As[0][0][0]),
                         GTM, GTK, (uint32_t)K);
        tdm_load_2d_bf16(W + (size_t)tileN * K, lds_off(&Bs[0][0][0]),
                         GTN, GTK, (uint32_t)K);
    }
#endif

    for (int it = 0; it < nIter; ++it) {
        const int k0 = it * GTK;
#if USE_TDM
        const int cur = it & 1;
        if (wave == 0) {
            if (it + 1 < nIter) {
                // prefetch next K-slab into the other buffer, then release
                // the current one (TDM completes in issue order per wave)
                tdm_load_2d_bf16(A + (size_t)tileM * K + (k0 + GTK),
                                 lds_off(&As[cur ^ 1][0][0]), GTM, GTK,
                                 (uint32_t)K);
                tdm_load_2d_bf16(W + (size_t)tileN * K + (k0 + GTK),
                                 lds_off(&Bs[cur ^ 1][0][0]), GTN, GTK,
                                 (uint32_t)K);
                __builtin_amdgcn_s_wait_tensorcnt(2);
            } else {
                __builtin_amdgcn_s_wait_tensorcnt(0);
            }
        }
        __syncthreads();
#else
        const int cur = 0;
        // A/B tiles: 128 rows x 64B = 4 x 16B chunks/row, 2 passes each
#pragma unroll
        for (int p = 0; p < 2; ++p) {
            int r = p * 64 + (tid >> 2);
            int c = tid & 3;
            *(uint4*)&As[0][r][c * 8] =
                *(const uint4*)(A + (size_t)(tileM + r) * K + k0 + c * 8);
            *(uint4*)&Bs[0][r][c * 8] =
                *(const uint4*)(W + (size_t)(tileN + r) * K + k0 + c * 8);
        }
        __syncthreads();
#endif

        v16bf af[2], bfv[4];
#pragma unroll
        for (int mi = 0; mi < 2; ++mi) {
            int row = wm * 32 + mi * 16 + l16;
            FragU u;
            u.q[0] = *(const uint4*)&As[cur][row][half * 8];
            u.q[1] = *(const uint4*)&As[cur][row][16 + half * 8];
            af[mi] = u.v;
        }
#pragma unroll
        for (int ni = 0; ni < 4; ++ni) {
            int row = wn * 64 + ni * 16 + l16;
            FragU u;
            u.q[0] = *(const uint4*)&Bs[cur][row][half * 8];
            u.q[1] = *(const uint4*)&Bs[cur][row][16 + half * 8];
            bfv[ni] = u.v;
        }
#pragma unroll
        for (int mi = 0; mi < 2; ++mi)
#pragma unroll
            for (int ni = 0; ni < 4; ++ni)
                acc[mi][ni] = __builtin_amdgcn_wmma_f32_16x16x32_bf16(
                    false, af[mi], false, bfv[ni], (short)0, acc[mi][ni],
                    false, false);
        __syncthreads();
    }

    // Epilogue: C layout: VGPR r holds row 8*half + r, col = l16
#pragma unroll
    for (int mi = 0; mi < 2; ++mi) {
#pragma unroll
        for (int ni = 0; ni < 4; ++ni) {
            int col = tileN + wn * 64 + ni * 16 + l16;
            float bv = bias[col];
#pragma unroll
            for (int r = 0; r < 8; ++r) {
                int row = tileM + wm * 32 + mi * 16 + half * 8 + r;
                float v = acc[mi][ni][r] + bv;
                if (out_is_bf16)
                    ((__bf16*)Cout)[(size_t)row * N + col] = f2bf(v);
                else
                    ((float*)Cout)[(size_t)row * N + col] = v;
            }
        }
    }
}

// ---------------------------------------------------------------------------
// Flash attention, Dh = 64.  Q/K/V in [B*S, E] bf16, head slice at h*64.
// Block = 4 waves; wave w owns q rows [q0 + 16w, q0 + 16w + 16).
// KV loop in 64-key tiles.  Logits scaled by 1/64 (reference uses 1/head_dim).
// P restaged through per-wave LDS to form WMMA A fragments.
// ---------------------------------------------------------------------------
__global__ __launch_bounds__(128) void flash_attn64(
    const __bf16* __restrict__ Q, const __bf16* __restrict__ K,
    const __bf16* __restrict__ V, __bf16* __restrict__ O,
    int S, int E)
{
    __shared__ __bf16 Kt[64][64];        // 8 KB  [kv][d]
    __shared__ __bf16 Vt[64][64];        // 8 KB  [d][kv] (transposed)
    __shared__ __bf16 Pb[4][16][64];     // 8 KB  per-wave P staging

    const int tid  = threadIdx.x;
    const int lane = tid & 31;
    const int wave = tid >> 5;           // 0..3
    const int half = lane >> 4;
    const int l16  = lane & 15;

    const int b  = blockIdx.z;
    const int h  = blockIdx.y;
    const int q0 = blockIdx.x * 64;
    const size_t baseBH = ((size_t)b * S) * E + (size_t)h * 64;

    // Preload Q A-fragments (16 rows x 64 d = two 16x32 frags per wave)
    v16bf qa[2];
    {
        int s = q0 + wave * 16 + l16;
        const __bf16* qp = Q + baseBH + (size_t)s * E;
#pragma unroll
        for (int kc = 0; kc < 2; ++kc) {
            FragU u;
            u.q[0] = *(const uint4*)(qp + kc * 32 + half * 8);
            u.q[1] = *(const uint4*)(qp + kc * 32 + 16 + half * 8);
            qa[kc] = u.v;
        }
    }

    v8f acc[4] = {};
    float mrow[8], lrow[8];
#pragma unroll
    for (int r = 0; r < 8; ++r) { mrow[r] = -1e30f; lrow[r] = 0.0f; }

    const float scale = 1.0f / 64.0f;

    for (int kv0 = 0; kv0 < S; kv0 += 64) {
        __syncthreads();   // previous iteration done reading Kt/Vt
        // K tile: 64 rows x 128B = 512 x 16B chunks / 128 threads
#pragma unroll
        for (int p = 0; p < 4; ++p) {
            int cidx = p * 128 + tid;
            int r = cidx >> 3, c = cidx & 7;
            *(uint4*)&Kt[r][c * 8] =
                *(const uint4*)(K + baseBH + (size_t)(kv0 + r) * E + c * 8);
        }
        // V tile with transpose into Vt[d][kv]
#pragma unroll
        for (int p = 0; p < 4; ++p) {
            int cidx = p * 128 + tid;
            int r = cidx >> 3, c = cidx & 7;
            uint4 raw =
                *(const uint4*)(V + baseBH + (size_t)(kv0 + r) * E + c * 8);
            const __bf16* e = (const __bf16*)&raw;
#pragma unroll
            for (int j = 0; j < 8; ++j) Vt[c * 8 + j][r] = e[j];
        }
        __syncthreads();

        // S = Q K^T  (contraction over d=64, two x32 WMMA steps)
        v8f st[4];
#pragma unroll
        for (int ni = 0; ni < 4; ++ni) {
            v8f z = {};
#pragma unroll
            for (int kc = 0; kc < 2; ++kc) {
                int row = ni * 16 + l16;   // kv index within tile
                FragU u;
                u.q[0] = *(const uint4*)&Kt[row][kc * 32 + half * 8];
                u.q[1] = *(const uint4*)&Kt[row][kc * 32 + 16 + half * 8];
                z = __builtin_amdgcn_wmma_f32_16x16x32_bf16(
                        false, qa[kc], false, u.v, (short)0, z, false, false);
            }
            st[ni] = z;
        }

        // Online softmax: row stats via 16-lane xor-shuffle reductions
#pragma unroll
        for (int r = 0; r < 8; ++r) {
            float lm = -1e30f;
#pragma unroll
            for (int ni = 0; ni < 4; ++ni) {
                st[ni][r] *= scale;
                lm = fmaxf(lm, st[ni][r]);
            }
#pragma unroll
            for (int off = 8; off >= 1; off >>= 1)
                lm = fmaxf(lm, __shfl_xor(lm, off, 32));
            float mnew  = fmaxf(mrow[r], lm);
            float alpha = __expf(mrow[r] - mnew);
            float rs = 0.0f;
#pragma unroll
            for (int ni = 0; ni < 4; ++ni) {
                float p = __expf(st[ni][r] - mnew);
                st[ni][r] = p;
                rs += p;
            }
#pragma unroll
            for (int off = 8; off >= 1; off >>= 1)
                rs += __shfl_xor(rs, off, 32);
            lrow[r] = lrow[r] * alpha + rs;
            mrow[r] = mnew;
#pragma unroll
            for (int ni = 0; ni < 4; ++ni) acc[ni][r] *= alpha;
        }

        // Stage P (C-layout -> LDS) then reload as A fragments.
#pragma unroll
        for (int ni = 0; ni < 4; ++ni)
#pragma unroll
            for (int r = 0; r < 8; ++r)
                Pb[wave][half * 8 + r][ni * 16 + l16] = f2bf(st[ni][r]);
        // Same-wave LDS RAW: DS ops are in-order per wave; wait + fence.
        asm volatile("s_wait_dscnt 0" ::: "memory");

        v16bf pa[2];
#pragma unroll
        for (int kc = 0; kc < 2; ++kc) {
            FragU u;
            u.q[0] = *(const uint4*)&Pb[wave][l16][kc * 32 + half * 8];
            u.q[1] = *(const uint4*)&Pb[wave][l16][kc * 32 + 16 + half * 8];
            pa[kc] = u.v;
        }
        // acc += P V   (contraction over kv=64)
#pragma unroll
        for (int ni = 0; ni < 4; ++ni) {
#pragma unroll
            for (int kc = 0; kc < 2; ++kc) {
                int row = ni * 16 + l16;   // d index
                FragU u;
                u.q[0] = *(const uint4*)&Vt[row][kc * 32 + half * 8];
                u.q[1] = *(const uint4*)&Vt[row][kc * 32 + 16 + half * 8];
                acc[ni] = __builtin_amdgcn_wmma_f32_16x16x32_bf16(
                    false, pa[kc], false, u.v, (short)0, acc[ni], false, false);
            }
        }
    }

    // Normalize and scatter to attn buffer [B*S, E] (bf16)
#pragma unroll
    for (int ni = 0; ni < 4; ++ni) {
        int col = h * 64 + ni * 16 + l16;
#pragma unroll
        for (int r = 0; r < 8; ++r) {
            int s = q0 + wave * 16 + half * 8 + r;
            float v = acc[ni][r] / lrow[r];
            O[((size_t)b * S + s) * E + col] = f2bf(v);
        }
    }
}

// ---------------------------------------------------------------------------
extern "C" void kernel_launch(void* const* d_in, const int* in_sizes, int n_in,
                              void* d_out, int out_size, void* d_ws, size_t ws_size,
                              hipStream_t stream)
{
    const float* x  = (const float*)d_in[0];
    const float* Wq = (const float*)d_in[1];
    const float* bq = (const float*)d_in[2];
    const float* Wk = (const float*)d_in[3];
    const float* bk = (const float*)d_in[4];
    const float* Wv = (const float*)d_in[5];
    const float* bv = (const float*)d_in[6];
    const float* Wo = (const float*)d_in[7];
    const float* bo = (const float*)d_in[8];
    float* out = (float*)d_out;

    const int B = 2, S = 2048, E = 1024, H = 16;
    const size_t M = (size_t)B * S;            // 4096 rows

    char* ws = (char*)d_ws;
    __bf16* xb  = (__bf16*)ws;  ws += M * E * sizeof(__bf16);          // 8 MB
    __bf16* wqb = (__bf16*)ws;  ws += (size_t)E * E * sizeof(__bf16);  // 2 MB
    __bf16* wkb = (__bf16*)ws;  ws += (size_t)E * E * sizeof(__bf16);
    __bf16* wvb = (__bf16*)ws;  ws += (size_t)E * E * sizeof(__bf16);
    __bf16* wob = (__bf16*)ws;  ws += (size_t)E * E * sizeof(__bf16);
    __bf16* Qb  = (__bf16*)ws;  ws += M * E * sizeof(__bf16);          // 8 MB
    __bf16* Kb  = (__bf16*)ws;  ws += M * E * sizeof(__bf16);
    __bf16* Vb  = (__bf16*)ws;  ws += M * E * sizeof(__bf16);
    __bf16* Ab  = (__bf16*)ws;  ws += M * E * sizeof(__bf16);
    (void)ws_size; (void)in_sizes; (void)n_in; (void)out_size;

    // 1) f32 -> bf16 conversions
    {
        int nx = (int)(M * E);
        cvt_f32_bf16<<<(nx / 4 + 255) / 256, 256, 0, stream>>>(x, xb, nx);
        int nw = E * E;
        cvt_f32_bf16<<<(nw / 4 + 255) / 256, 256, 0, stream>>>(Wq, wqb, nw);
        cvt_f32_bf16<<<(nw / 4 + 255) / 256, 256, 0, stream>>>(Wk, wkb, nw);
        cvt_f32_bf16<<<(nw / 4 + 255) / 256, 256, 0, stream>>>(Wv, wvb, nw);
        cvt_f32_bf16<<<(nw / 4 + 255) / 256, 256, 0, stream>>>(Wo, wob, nw);
    }

    // 2) Q/K/V projections (bf16 out)
    dim3 gg((unsigned)(M / GTM), (unsigned)(E / GTN));
    wmma_gemm_bf16<<<gg, 256, 0, stream>>>(xb, wqb, bq, Qb, (int)M, E, E, 1);
    wmma_gemm_bf16<<<gg, 256, 0, stream>>>(xb, wkb, bk, Kb, (int)M, E, E, 1);
    wmma_gemm_bf16<<<gg, 256, 0, stream>>>(xb, wvb, bv, Vb, (int)M, E, E, 1);

    // 3) flash attention
    flash_attn64<<<dim3(S / 64, H, B), 128, 0, stream>>>(Qb, Kb, Vb, Ab, S, E);

    // 4) output projection (fp32 out + bias)
    wmma_gemm_bf16<<<gg, 256, 0, stream>>>(Ab, wob, bo, out, (int)M, E, E, 0);
}
